// ProfiledNextCharBlock_41369124995422
// MI455X (gfx1250) — matrix operand order repaired and checked by reading, not compiled, not run on hardware
//
#include <hip/hip_runtime.h>
#include <hip/hip_bf16.h>

// ---------------------------------------------------------------------------
// Mamba-2-style block for MI455X (gfx1250, wave32, WMMA).
// GEMMs: v_wmma_f32_16x16x32_bf16, fp32 accum, LDS-staged double-buffered B
// tiles via async global->LDS copies. SSM scan: chunked 3-pass parallel scan.
// ---------------------------------------------------------------------------

#define D_MODEL   1024
#define D_INNER   2048
#define N_HEADS   32
#define D_HEAD    64
#define D_STATE   16
#define D_CONV    4
#define PARAM_DIM (2 * N_HEADS)                            // 64
#define PROJ_OUT  (2 * D_INNER + PARAM_DIM + 4 * D_STATE)  // 4224
#define D_FF      (4 * D_MODEL)                            // 4096
#define SEQ_T     2048
#define SCAN_CH   16                                        // time chunks

typedef __attribute__((ext_vector_type(16))) __bf16 v16bf;
typedef __attribute__((ext_vector_type(8)))  float  v8f;
typedef int vint4 __attribute__((vector_size(16)));

#if defined(__HIP_DEVICE_COMPILE__) && \
    __has_builtin(__builtin_amdgcn_global_load_async_to_lds_b128) && \
    __has_builtin(__builtin_amdgcn_s_wait_asynccnt)
#define HAVE_ASYNC_LDS 1
#else
#define HAVE_ASYNC_LDS 0
#endif

// ---------------------- scalar helpers ----------------------

__device__ __forceinline__ unsigned short f2bf(float f) {
    unsigned u = __builtin_bit_cast(unsigned, f);
    unsigned r = u + 0x7FFFu + ((u >> 16) & 1u);   // round-to-nearest-even
    return (unsigned short)(r >> 16);
}

__device__ __forceinline__ float siluf(float x)     { return x / (1.0f + __expf(-x)); }
__device__ __forceinline__ float softplusf(float x) { return (x > 20.0f) ? x : log1pf(__expf(x)); }

__device__ __forceinline__ float geluf(float x) {
    float x3 = x * x * x;
    return 0.5f * x * (1.0f + tanhf(0.7978845608028654f * (x + 0.044715f * x3)));
}

// ---------------------- async global -> LDS staging ----------------------

__device__ __forceinline__ void stage_b128(const unsigned short* gsrc, unsigned short* ldst) {
#if HAVE_ASYNC_LDS
    __builtin_amdgcn_global_load_async_to_lds_b128(
        (vint4 __attribute__((address_space(1)))*)(gsrc),
        (vint4 __attribute__((address_space(3)))*)(ldst), 0, 0);
#else
    *(uint4*)ldst = *(const uint4*)gsrc;   // sync fallback (still LDS-staged)
#endif
}

__device__ __forceinline__ void stage_wait() {
#if HAVE_ASYNC_LDS
    __builtin_amdgcn_s_wait_asynccnt(0);
#endif
}

// ---------------------- WMMA fragment loads ----------------------
// A matrix 16x32 bf16 (MxK), contiguous k order (matches B):
//   lanes 0-15 : M = lane,    K = 0..15
//   lanes 16-31: M = lane-16, K = 16..31
__device__ __forceinline__ v16bf load_frag_a(const unsigned short* A, int lda,
                                             int m0, int k0, int lane) {
    int m  = m0 + (lane & 15);
    int ko = k0 + ((lane & 16) ? 16 : 0);
    const unsigned short* p = A + (size_t)m * lda + ko;
    union { unsigned short u[16]; v16bf v; } f;
    *(uint4*)(f.u)     = *(const uint4*)(p);
    *(uint4*)(f.u + 8) = *(const uint4*)(p + 8);
    return f.v;
}

// B matrix 32x16 (KxN) read from an LDS tile laid out as [n_local][32 k] bf16:
//   lanes 0-15 : N = lane,    K = 0..15
//   lanes 16-31: N = lane-16, K = 16..31
__device__ __forceinline__ v16bf load_frag_b_lds(const unsigned short* tile,
                                                 int nloc, int lane) {
    int n  = nloc + (lane & 15);
    int ko = (lane & 16) ? 16 : 0;
    const unsigned short* p = tile + n * 32 + ko;
    union { unsigned short u[16]; v16bf v; } f;
    *(uint4*)(f.u)     = *(const uint4*)(p);        // ds_load_b128
    *(uint4*)(f.u + 8) = *(const uint4*)(p + 8);
    return f.v;
}

// ---------------------- weight conversion ----------------------

__global__ void f32_to_bf16_kernel(const float* __restrict__ in,
                                   unsigned short* __restrict__ out, size_t n) {
    size_t i = (size_t)blockIdx.x * blockDim.x + threadIdx.x;
    if (i < n) out[i] = f2bf(in[i]);
}

// ---------------------- rmsnorm -> bf16 ----------------------

__global__ void rmsnorm_bf16_kernel(const float* __restrict__ x,
                                    const float* __restrict__ w,
                                    unsigned short* __restrict__ out, int D) {
    int row = blockIdx.x;
    const float* xr = x + (size_t)row * D;
    __shared__ float red[9];
    float s = 0.0f;
    for (int i = threadIdx.x; i < D; i += blockDim.x) { float v = xr[i]; s += v * v; }
    for (int off = 16; off > 0; off >>= 1) s += __shfl_down(s, off, 32);
    if ((threadIdx.x & 31) == 0) red[threadIdx.x >> 5] = s;
    __syncthreads();
    if (threadIdx.x == 0) {
        float t = 0.0f;
        int nw = blockDim.x >> 5;
        for (int i = 0; i < nw; ++i) t += red[i];
        red[8] = rsqrtf(t / (float)D + 1e-6f);
    }
    __syncthreads();
    float rs = red[8];
    for (int i = threadIdx.x; i < D; i += blockDim.x)
        out[(size_t)row * D + i] = f2bf(xr[i] * rs * w[i]);
}

// ---------------------- gated rmsnorm (scan_y * silu(gate)) ----------------------

__global__ void gated_rmsnorm_kernel(const float* __restrict__ y,
                                     const float* __restrict__ proj,
                                     const float* __restrict__ w,
                                     unsigned short* __restrict__ out) {
    const int D = D_INNER;
    int row = blockIdx.x;
    const float* yr = y + (size_t)row * D;
    const float* gr = proj + (size_t)row * PROJ_OUT;   // gate = proj[..., :D_INNER]
    __shared__ float zbuf[D_INNER];
    __shared__ float red[9];
    float s = 0.0f;
    for (int i = threadIdx.x; i < D; i += blockDim.x) {
        float z = yr[i] * siluf(gr[i]);
        zbuf[i] = z;
        s += z * z;
    }
    for (int off = 16; off > 0; off >>= 1) s += __shfl_down(s, off, 32);
    if ((threadIdx.x & 31) == 0) red[threadIdx.x >> 5] = s;
    __syncthreads();
    if (threadIdx.x == 0) {
        float t = 0.0f;
        int nw = blockDim.x >> 5;
        for (int i = 0; i < nw; ++i) t += red[i];
        red[8] = rsqrtf(t / (float)D + 1e-6f);
    }
    __syncthreads();
    float rs = red[8];
    for (int i = threadIdx.x; i < D; i += blockDim.x)
        out[(size_t)row * D + i] = f2bf(zbuf[i] * rs * w[i]);
}

// ---------------------- depthwise causal conv + SiLU ----------------------

__global__ void conv_silu_kernel(const float* __restrict__ proj,
                                 const float* __restrict__ conv_w,
                                 const float* __restrict__ conv_b,
                                 float* __restrict__ value, int T, size_t total) {
    size_t idx = (size_t)blockIdx.x * blockDim.x + threadIdx.x;
    if (idx >= total) return;
    int c  = (int)(idx % D_INNER);
    size_t bt = idx / D_INNER;
    int t = (int)(bt % T);
    int b = (int)(bt / T);
    float s = conv_b[c];
    #pragma unroll
    for (int k = 0; k < D_CONV; ++k) {
        int tt = t - (D_CONV - 1) + k;
        if (tt >= 0)
            s += proj[((size_t)b * T + tt) * PROJ_OUT + D_INNER + c] * conv_w[c * D_CONV + k];
    }
    value[idx] = siluf(s);
}

// ---------------------- chunked SSM scan ----------------------
// Recurrence per (b,h,p,n): S' = M_t S + add_t; M_t (2x2) shared per head.
// Chunk map is affine: S_out = P_c S_in + q_c with P_c head-shared.
// pass1: each (b,h,chunk) runs from zero state -> q_c (full state) and P_c.
// pass2: per (b,h), 16-step sequential combine -> true initial state per chunk.
// pass3: each (b,h,chunk) replays from its true init, writing y in place.

__device__ __forceinline__ void scan_load_step(const float* pr, int h, int p,
                                               float sB[2][D_STATE],
                                               float sC[2][D_STATE], float sP[2]) {
    if (p < 32) {
        sB[p >> 4][p & 15] = pr[2 * D_INNER + PARAM_DIM + p];
    } else {
        int q = p - 32;
        sC[q >> 4][q & 15] = pr[2 * D_INNER + PARAM_DIM + 32 + q];
    }
    if (p == 0) sP[0] = softplusf(pr[2 * D_INNER + h * 2 + 0]);
    if (p == 1) sP[1] = 1.0f / (1.0f + __expf(-pr[2 * D_INNER + h * 2 + 1]));
}

__global__ void scan_chunk_pass1(const float* __restrict__ proj,
                                 const float* __restrict__ value,
                                 float* __restrict__ qbuf,
                                 float* __restrict__ Pbuf, int T) {
    int idx = blockIdx.x;                 // ((b*H + h)*CH + c)
    int c   = idx % SCAN_CH;
    int bh  = idx / SCAN_CH;
    int h   = bh % N_HEADS;
    int b   = bh / N_HEADS;
    int p   = threadIdx.x;                // 0..63
    int Tc  = T / SCAN_CH;
    int t0  = c * Tc;

    __shared__ float sB[2][D_STATE], sC[2][D_STATE], sP[2];
    float S[D_STATE][2];
    #pragma unroll
    for (int n = 0; n < D_STATE; ++n) { S[n][0] = 0.0f; S[n][1] = 0.0f; }
    float P00 = 1.0f, P01 = 0.0f, P10 = 0.0f, P11 = 1.0f;

    for (int tt = 0; tt < Tc; ++tt) {
        size_t row = (size_t)b * T + (t0 + tt);
        scan_load_step(proj + row * PROJ_OUT, h, p, sB, sC, sP);
        __syncthreads();
        float A = sP[0], dt = sP[1];
        float u = value[row * D_INNER + h * D_HEAD + p];
        float K0 = dt, K1 = dt * dt;
        float m01 = -dt * A, m10 = dt, m11 = 1.0f - dt * dt * A;
        #pragma unroll
        for (int n = 0; n < D_STATE; ++n) {
            float add0 = K0 * sB[0][n] * u;
            float add1 = K1 * sB[1][n] * u;
            float s0 = S[n][0], s1 = S[n][1];
            S[n][0] = s0 + m01 * s1 + add0;
            S[n][1] = m10 * s0 + m11 * s1 + add1;
        }
        // P = M_t * P (head-shared; computed redundantly by all lanes)
        float n00 = P00 + m01 * P10;
        float n01 = P01 + m01 * P11;
        float n10 = m10 * P00 + m11 * P10;
        float n11 = m10 * P01 + m11 * P11;
        P00 = n00; P01 = n01; P10 = n10; P11 = n11;
        __syncthreads();
    }

    size_t qb = (size_t)idx * (D_HEAD * D_STATE * 2) + (size_t)p * (D_STATE * 2);
    #pragma unroll
    for (int n = 0; n < D_STATE; ++n) {
        qbuf[qb + n * 2 + 0] = S[n][0];
        qbuf[qb + n * 2 + 1] = S[n][1];
    }
    if (p == 0) {
        Pbuf[idx * 4 + 0] = P00; Pbuf[idx * 4 + 1] = P01;
        Pbuf[idx * 4 + 2] = P10; Pbuf[idx * 4 + 3] = P11;
    }
}

__global__ void scan_chunk_pass2(float* __restrict__ qbuf,
                                 const float* __restrict__ Pbuf) {
    int bh = blockIdx.x;      // b*H + h
    int p  = threadIdx.x;     // 0..63
    float S[D_STATE][2];
    #pragma unroll
    for (int n = 0; n < D_STATE; ++n) { S[n][0] = 0.0f; S[n][1] = 0.0f; }

    for (int c = 0; c < SCAN_CH; ++c) {
        int idx = bh * SCAN_CH + c;
        size_t qb = (size_t)idx * (D_HEAD * D_STATE * 2) + (size_t)p * (D_STATE * 2);
        float P00 = Pbuf[idx * 4 + 0], P01 = Pbuf[idx * 4 + 1];
        float P10 = Pbuf[idx * 4 + 2], P11 = Pbuf[idx * 4 + 3];
        #pragma unroll
        for (int n = 0; n < D_STATE; ++n) {
            float q0 = qbuf[qb + n * 2 + 0];
            float q1 = qbuf[qb + n * 2 + 1];
            // publish true initial state for this chunk
            qbuf[qb + n * 2 + 0] = S[n][0];
            qbuf[qb + n * 2 + 1] = S[n][1];
            // advance running state: S = P_c S + q_c
            float s0 = S[n][0], s1 = S[n][1];
            S[n][0] = P00 * s0 + P01 * s1 + q0;
            S[n][1] = P10 * s0 + P11 * s1 + q1;
        }
    }
}

__global__ void scan_chunk_pass3(const float* __restrict__ proj,
                                 float* __restrict__ value,
                                 const float* __restrict__ qbuf, int T) {
    int idx = blockIdx.x;
    int c   = idx % SCAN_CH;
    int bh  = idx / SCAN_CH;
    int h   = bh % N_HEADS;
    int b   = bh / N_HEADS;
    int p   = threadIdx.x;
    int Tc  = T / SCAN_CH;
    int t0  = c * Tc;

    __shared__ float sB[2][D_STATE], sC[2][D_STATE], sP[2];
    float S[D_STATE][2];
    size_t qb = (size_t)idx * (D_HEAD * D_STATE * 2) + (size_t)p * (D_STATE * 2);
    #pragma unroll
    for (int n = 0; n < D_STATE; ++n) {
        S[n][0] = qbuf[qb + n * 2 + 0];
        S[n][1] = qbuf[qb + n * 2 + 1];
    }

    for (int tt = 0; tt < Tc; ++tt) {
        size_t row = (size_t)b * T + (t0 + tt);
        scan_load_step(proj + row * PROJ_OUT, h, p, sB, sC, sP);
        __syncthreads();
        float A = sP[0], dt = sP[1];
        float u = value[row * D_INNER + h * D_HEAD + p];
        float K0 = dt, K1 = dt * dt;
        float m01 = -dt * A, m10 = dt, m11 = 1.0f - dt * dt * A;
        float y = 0.0f;
        #pragma unroll
        for (int n = 0; n < D_STATE; ++n) {
            float add0 = K0 * sB[0][n] * u;
            float add1 = K1 * sB[1][n] * u;
            float s0 = S[n][0], s1 = S[n][1];
            float ns0 = s0 + m01 * s1 + add0;
            float ns1 = m10 * s0 + m11 * s1 + add1;
            S[n][0] = ns0; S[n][1] = ns1;
            y += sC[0][n] * ns0 + sC[1][n] * ns1;
        }
        value[row * D_INNER + h * D_HEAD + p] = y;   // overwrite conv value with y
        __syncthreads();
    }
}

// ---------------------- WMMA GEMM: C(MxN) = A(MxK) . W(NxK)^T ----------------------
// 8 waves/block; block tile 128 rows x 64 cols. B strip (shared by all waves)
// is staged in LDS with double-buffered async global->LDS copies.
// Epilogues: 0 raw f32 | 1 +residual f32 | 2 +bias,GELU -> bf16 | 3 +bias+residual f32
template <int EPI>
__global__ void gemm_bf16_wmma(const unsigned short* __restrict__ A,
                               const unsigned short* __restrict__ W,
                               int M, int N, int K,
                               float* __restrict__ Cf,
                               unsigned short* __restrict__ Cb,
                               const float* __restrict__ bias,
                               const float* __restrict__ residual) {
    __shared__ alignas(16) unsigned short Bt[2][64 * 32];   // [buf][n_local*32 + k]

    int lane = threadIdx.x & 31;
    int wave = threadIdx.x >> 5;
    int tid  = threadIdx.x;
    int m0 = blockIdx.y * 128 + wave * 16;
    int n0 = blockIdx.x * 64;

    // cooperative B staging: thread -> (row r = tid/4, 16B chunk c = tid%4)
    int br = tid >> 2;
    int bc = (tid & 3) * 8;                     // ushort offset within 32-k row
    const unsigned short* gB = W + (size_t)(n0 + br) * K + bc;
    unsigned short* l0 = &Bt[0][br * 32 + bc];
    unsigned short* l1 = &Bt[1][br * 32 + bc];

    stage_b128(gB, l0);                          // preload k-tile 0

    v8f acc0 = (v8f)0.0f, acc1 = (v8f)0.0f, acc2 = (v8f)0.0f, acc3 = (v8f)0.0f;

    int nkt = K >> 5;
    for (int kt = 0; kt < nkt; ++kt) {
        int cur = kt & 1;
        stage_wait();
        __syncthreads();                         // tile `cur` ready; prior reads done
        if (kt + 1 < nkt)
            stage_b128(gB + (size_t)(kt + 1) * 32, cur ? l0 : l1);

        v16bf a = load_frag_a(A, K, m0, kt * 32, lane);
        if (kt + 1 < nkt)
            __builtin_prefetch(A + (size_t)(m0 + (lane & 15)) * K + (kt + 1) * 32, 0, 0);

        const unsigned short* tile = Bt[cur];
        v16bf b0 = load_frag_b_lds(tile,  0, lane);
        acc0 = __builtin_amdgcn_wmma_f32_16x16x32_bf16(false, a, false, b0, (short)0, acc0, false, false);
        v16bf b1 = load_frag_b_lds(tile, 16, lane);
        acc1 = __builtin_amdgcn_wmma_f32_16x16x32_bf16(false, a, false, b1, (short)0, acc1, false, false);
        v16bf b2 = load_frag_b_lds(tile, 32, lane);
        acc2 = __builtin_amdgcn_wmma_f32_16x16x32_bf16(false, a, false, b2, (short)0, acc2, false, false);
        v16bf b3 = load_frag_b_lds(tile, 48, lane);
        acc3 = __builtin_amdgcn_wmma_f32_16x16x32_bf16(false, a, false, b3, (short)0, acc3, false, false);
    }

    // C layout: VGPR v -> row m0 + v (+8 for lanes 16-31), col n0 + (lane&15)
    int rowBase = m0 + ((lane & 16) ? 8 : 0);
    int colBase = n0 + (lane & 15);
    v8f accs[4] = {acc0, acc1, acc2, acc3};
    #pragma unroll
    for (int tcol = 0; tcol < 4; ++tcol) {
        int n = colBase + 16 * tcol;
        #pragma unroll
        for (int v = 0; v < 8; ++v) {
            int m = rowBase + v;
            size_t idx = (size_t)m * N + n;
            float val = accs[tcol][v];
            if constexpr (EPI == 0) {
                Cf[idx] = val;
            } else if constexpr (EPI == 1) {
                Cf[idx] = val + residual[idx];
            } else if constexpr (EPI == 2) {
                Cb[idx] = f2bf(geluf(val + bias[n]));
            } else {
                Cf[idx] = val + bias[n] + residual[idx];
            }
        }
    }
}

// ---------------------- host launcher ----------------------

extern "C" void kernel_launch(void* const* d_in, const int* in_sizes, int n_in,
                              void* d_out, int out_size, void* d_ws, size_t ws_size,
                              hipStream_t stream) {
    const float* x         = (const float*)d_in[0];
    const float* w_in      = (const float*)d_in[1];
    const float* conv_w    = (const float*)d_in[2];
    const float* conv_b    = (const float*)d_in[3];
    const float* norm1_w   = (const float*)d_in[4];
    const float* outnorm_w = (const float*)d_in[5];
    const float* w_out     = (const float*)d_in[6];
    const float* norm2_w   = (const float*)d_in[7];
    const float* ff_w1     = (const float*)d_in[8];
    const float* ff_b1     = (const float*)d_in[9];
    const float* ff_w2     = (const float*)d_in[10];
    const float* ff_b2     = (const float*)d_in[11];
    float* out = (float*)d_out;

    const int BT = in_sizes[0] / D_MODEL;   // B*T = 4096
    const int T  = SEQ_T;
    const int B  = BT / T;

    char* ws = (char*)d_ws;
    size_t off = 0;
    auto carve = [&](size_t bytes) { char* p = ws + off; off += (bytes + 255) & ~(size_t)255; return p; };

    unsigned short* win_bf  = (unsigned short*)carve((size_t)PROJ_OUT * D_MODEL * 2);
    unsigned short* wout_bf = (unsigned short*)carve((size_t)D_MODEL * D_INNER * 2);
    unsigned short* ffw1_bf = (unsigned short*)carve((size_t)D_FF * D_MODEL * 2);
    unsigned short* ffw2_bf = (unsigned short*)carve((size_t)D_MODEL * D_FF * 2);
    unsigned short* h_bf    = (unsigned short*)carve((size_t)BT * D_MODEL * 2);
    float*          proj    = (float*)carve((size_t)BT * PROJ_OUT * 4);
    float*          value   = (float*)carve((size_t)BT * D_INNER * 4);   // conv out -> y (in place)
    unsigned short* yg_bf   = (unsigned short*)carve((size_t)BT * D_INNER * 2);
    float*          x1      = (float*)carve((size_t)BT * D_MODEL * 4);
    unsigned short* h2_bf   = (unsigned short*)carve((size_t)BT * D_MODEL * 2);
    unsigned short* ffm_bf  = (unsigned short*)carve((size_t)BT * D_FF * 2);
    float*          qbuf    = (float*)carve((size_t)B * N_HEADS * SCAN_CH * D_HEAD * D_STATE * 2 * 4);
    float*          Pbuf    = (float*)carve((size_t)B * N_HEADS * SCAN_CH * 4 * 4);

    auto cdiv = [](size_t a, size_t b) { return (unsigned)((a + b - 1) / b); };

    // 1. weights -> bf16
    {
        size_t n1 = (size_t)PROJ_OUT * D_MODEL;
        size_t n2 = (size_t)D_MODEL * D_INNER;
        size_t n3 = (size_t)D_FF * D_MODEL;
        size_t n4 = (size_t)D_MODEL * D_FF;
        f32_to_bf16_kernel<<<cdiv(n1, 256), 256, 0, stream>>>(w_in,  win_bf,  n1);
        f32_to_bf16_kernel<<<cdiv(n2, 256), 256, 0, stream>>>(w_out, wout_bf, n2);
        f32_to_bf16_kernel<<<cdiv(n3, 256), 256, 0, stream>>>(ff_w1, ffw1_bf, n3);
        f32_to_bf16_kernel<<<cdiv(n4, 256), 256, 0, stream>>>(ff_w2, ffw2_bf, n4);
    }

    // 2. h = rmsnorm(x, norm1_w) -> bf16
    rmsnorm_bf16_kernel<<<BT, 256, 0, stream>>>(x, norm1_w, h_bf, D_MODEL);

    // 3. proj = h . w_in^T   (BT x 4224)
    gemm_bf16_wmma<0><<<dim3(PROJ_OUT / 64, BT / 128), 256, 0, stream>>>(
        h_bf, win_bf, BT, PROJ_OUT, D_MODEL, proj, nullptr, nullptr, nullptr);

    // 4. causal depthwise conv + SiLU -> value
    {
        size_t total = (size_t)BT * D_INNER;
        conv_silu_kernel<<<cdiv(total, 256), 256, 0, stream>>>(proj, conv_w, conv_b, value, T, total);
    }

    // 5. chunked SSM scan (3 passes), y written in place over `value`
    scan_chunk_pass1<<<B * N_HEADS * SCAN_CH, D_HEAD, 0, stream>>>(proj, value, qbuf, Pbuf, T);
    scan_chunk_pass2<<<B * N_HEADS, D_HEAD, 0, stream>>>(qbuf, Pbuf);
    scan_chunk_pass3<<<B * N_HEADS * SCAN_CH, D_HEAD, 0, stream>>>(proj, value, qbuf, T);

    // 6. yg = rmsnorm(y * silu(gate), out_norm_w) -> bf16
    gated_rmsnorm_kernel<<<BT, 256, 0, stream>>>(value, proj, outnorm_w, yg_bf);

    // 7. x1 = x + yg . w_out^T
    gemm_bf16_wmma<1><<<dim3(D_MODEL / 64, BT / 128), 256, 0, stream>>>(
        yg_bf, wout_bf, BT, D_MODEL, D_INNER, x1, nullptr, nullptr, x);

    // 8. h2 = rmsnorm(x1, norm2_w) -> bf16
    rmsnorm_bf16_kernel<<<BT, 256, 0, stream>>>(x1, norm2_w, h2_bf, D_MODEL);

    // 9. ffmid = gelu(h2 . ff_w1^T + b1) -> bf16
    gemm_bf16_wmma<2><<<dim3(D_FF / 64, BT / 128), 256, 0, stream>>>(
        h2_bf, ffw1_bf, BT, D_FF, D_MODEL, nullptr, ffm_bf, ff_b1, nullptr);

    // 10. out = x1 + ffmid . ff_w2^T + b2
    gemm_bf16_wmma<3><<<dim3(D_MODEL / 64, BT / 128), 256, 0, stream>>>(
        ffm_bf, ffw2_bf, BT, D_MODEL, D_FF, out, nullptr, ff_b2, x1);

    (void)n_in; (void)out_size; (void)ws_size;
}
